// TargetEdgeInitializer_22342419874266
// MI455X (gfx1250) — matrix operand order repaired and checked by reading, not compiled, not run on hardware
//
#include <hip/hip_runtime.h>
#include <hip/hip_bf16.h>
#include <stdint.h>

// Problem constants (from reference)
constexpr int N_ = 10000;
constexpr int E_ = 160000;
constexpr int HC = 512;   // H*C = D
constexpr int Kp = 10016; // N padded to multiple of 32 for WMMA K loops

typedef __attribute__((ext_vector_type(16))) __bf16 v16bf;
typedef __attribute__((ext_vector_type(8)))  float  v8f;

// ---------- helpers ----------
__device__ __forceinline__ unsigned short f2bf(float f) {
  unsigned u = __float_as_uint(f);
  unsigned r = (u + 0x7FFFu + ((u >> 16) & 1u)) >> 16; // round-to-nearest-even
  return (unsigned short)r;
}
// order-preserving float<->uint encoding (for atomic min/max on floats)
__device__ __forceinline__ unsigned f2ord(float f) {
  unsigned u = __float_as_uint(f);
  return (u & 0x80000000u) ? ~u : (u | 0x80000000u);
}
__device__ __forceinline__ float ord2f(unsigned o) {
  unsigned u = (o & 0x80000000u) ? (o & 0x7FFFFFFFu) : ~o;
  return __uint_as_float(u);
}

// ---------- f32 -> bf16 conversion (with zero padding) ----------
__global__ __launch_bounds__(256) void f32_to_bf16_kernel(
    const float* __restrict__ src, unsigned short* __restrict__ dst,
    int count, int srcCount) {
  int i = blockIdx.x * 256 + threadIdx.x;
  if (i < count) dst[i] = (i < srcCount) ? f2bf(src[i]) : (unsigned short)0;
}

// ---------- fused 4-way GEMM: C[sel] = A @ B[sel] + bias[sel] ----------
// A: bf16 [M x K] row-major (lda=K). B: bf16 [K x 512] row-major. C: f32 [M x 512].
// One wave computes a 16(M) x 64(N) tile; 8 waves/block cover all 512 columns.
struct Gemm4 {
  const unsigned short* B[4];
  const float* bias[4];
  float* C[4];
};

__global__ __launch_bounds__(256) void gemm4_bf16_kernel(
    const unsigned short* __restrict__ A, Gemm4 p, int K) {
  const int sel = blockIdx.y;
  const unsigned short* __restrict__ Bm = p.B[sel];
  const float* __restrict__ bias = p.bias[sel];
  float* __restrict__ Cm = p.C[sel];

  const int wave = threadIdx.x >> 5;
  const int lane = threadIdx.x & 31;
  const int m0 = blockIdx.x * 16;
  const int n0 = wave * 64;
  const int l15 = lane & 15;

  // A fragment (16x32 bf16): lane<16 holds row m0+lane, K = {0..7,16..23}; lane>=16 K = {8..15,24..31}
  const unsigned short* Arow = A + (size_t)(m0 + l15) * K + ((lane < 16) ? 0 : 8);
  // B fragment (32x16 bf16): lane holds column n0+l15, K = {0..7,16..23} (lane<16) / {8..15,24..31} (lane>=16)
  const unsigned short* Brow = Bm + (size_t)((lane < 16) ? 0 : 8) * HC + n0 + l15;

  v8f acc0{}, acc1{}, acc2{}, acc3{};
  for (int k0 = 0; k0 < K; k0 += 32) {
    union { uint32_t u[8]; v16bf v; } af;
#pragma unroll
    for (int i = 0; i < 4; ++i) {
      af.u[i]     = *(const uint32_t*)(Arow + 2 * i);
      af.u[4 + i] = *(const uint32_t*)(Arow + 16 + 2 * i);
    }
    union { unsigned short s[16]; v16bf v; } b0, b1, b2, b3;
#pragma unroll
    for (int i = 0; i < 8; ++i) {
      b0.s[i] = Brow[(size_t)i * HC +  0]; b0.s[8 + i] = Brow[(size_t)(16 + i) * HC +  0];
      b1.s[i] = Brow[(size_t)i * HC + 16]; b1.s[8 + i] = Brow[(size_t)(16 + i) * HC + 16];
      b2.s[i] = Brow[(size_t)i * HC + 32]; b2.s[8 + i] = Brow[(size_t)(16 + i) * HC + 32];
      b3.s[i] = Brow[(size_t)i * HC + 48]; b3.s[8 + i] = Brow[(size_t)(16 + i) * HC + 48];
    }
    acc0 = __builtin_amdgcn_wmma_f32_16x16x32_bf16(false, af.v, false, b0.v, (short)0, acc0, false, false);
    acc1 = __builtin_amdgcn_wmma_f32_16x16x32_bf16(false, af.v, false, b1.v, (short)0, acc1, false, false);
    acc2 = __builtin_amdgcn_wmma_f32_16x16x32_bf16(false, af.v, false, b2.v, (short)0, acc2, false, false);
    acc3 = __builtin_amdgcn_wmma_f32_16x16x32_bf16(false, af.v, false, b3.v, (short)0, acc3, false, false);
    Arow += 32;
    Brow += (size_t)32 * HC;
  }
  // C layout: VGPR r -> row m0+r (lane<16) / m0+8+r (lane>=16), col = n0 + (lane&15) + 16*t
  const int crow = m0 + ((lane < 16) ? 0 : 8);
  const int ccol = n0 + l15;
  float* Cp = Cm + (size_t)crow * HC + ccol;
#pragma unroll
  for (int r = 0; r < 8; ++r) {
    Cp[(size_t)r * HC +  0] = acc0[r] + bias[ccol +  0];
    Cp[(size_t)r * HC + 16] = acc1[r] + bias[ccol + 16];
    Cp[(size_t)r * HC + 32] = acc2[r] + bias[ccol + 32];
    Cp[(size_t)r * HC + 48] = acc3[r] + bias[ccol + 48];
  }
}

// ---------- gram: G = h^T h via WMMA, A fragment loaded transposed ----------
__global__ __launch_bounds__(256) void gram_bf16_kernel(
    const unsigned short* __restrict__ Hb, float* __restrict__ G, int Krows) {
  const int wave = threadIdx.x >> 5;
  const int lane = threadIdx.x & 31;
  const int tile = blockIdx.x * 8 + wave;   // 1024 tiles of 16x16
  const int m0 = (tile >> 5) * 16;
  const int n0 = (tile & 31) * 16;
  const int l15 = lane & 15;
  const int kb = (lane < 16) ? 0 : 8;
  v8f acc{};
  for (int k0 = 0; k0 < Krows; k0 += 32) {
    union { unsigned short s[16]; v16bf v; } af; // (h^T)[m0+l15, k...] = Hb[k, m0+l15]
    const unsigned short* ac = Hb + (size_t)(k0 + kb) * HC + (m0 + l15);
#pragma unroll
    for (int i = 0; i < 8; ++i) {
      af.s[i]     = ac[(size_t)i * HC];
      af.s[8 + i] = ac[(size_t)(16 + i) * HC];
    }
    union { unsigned short s[16]; v16bf v; } bfr;
    const unsigned short* bc = Hb + (size_t)(k0 + kb) * HC + (n0 + l15);
#pragma unroll
    for (int i = 0; i < 8; ++i) { bfr.s[i] = bc[(size_t)i * HC]; bfr.s[8 + i] = bc[(size_t)(16 + i) * HC]; }
    acc = __builtin_amdgcn_wmma_f32_16x16x32_bf16(false, af.v, false, bfr.v, (short)0, acc, false, false);
  }
  const int crow = m0 + ((lane < 16) ? 0 : 8);
#pragma unroll
  for (int r = 0; r < 8; ++r)
    G[(size_t)(crow + r) * HC + n0 + l15] = acc[r];
}

// ---------- edge pass 1: logits + segment max (one wave per (edge,head)) ----------
__global__ __launch_bounds__(256) void edge_logits_kernel(
    const float* __restrict__ q, const float* __restrict__ k,
    const int64_t* __restrict__ ei, const float* __restrict__ ea,
    const float* __restrict__ Wel, float* __restrict__ logits,
    unsigned* __restrict__ m_ord) {
  const int gw = (blockIdx.x * 256 + threadIdx.x) >> 5;
  const int lane = threadIdx.x & 31;
  const int e = gw >> 2;
  const int h = gw & 3;
  if (e >= E_) return;
  const int src = (int)ei[e];
  const int dst = (int)ei[E_ + e];
  const int co = h * 128 + lane * 4;
  const float4 qv = *(const float4*)(q + (size_t)dst * HC + co);
  const float4 kv = *(const float4*)(k + (size_t)src * HC + co);
  const float4 wv = *(const float4*)(Wel + co);
  const float ew = ea[e];
  float p = qv.x * (kv.x + ew * wv.x) + qv.y * (kv.y + ew * wv.y)
          + qv.z * (kv.z + ew * wv.z) + qv.w * (kv.w + ew * wv.w);
#pragma unroll
  for (int off = 16; off > 0; off >>= 1) p += __shfl_xor(p, off, 32);
  if (lane == 0) {
    const float lg = p * 0.08838834764831845f; // 1/sqrt(128)
    logits[(size_t)e * 4 + h] = lg;
    atomicMax(&m_ord[(size_t)dst * 4 + h], f2ord(lg));
  }
}

// ---------- edge pass 2: exp + denom ----------
__global__ __launch_bounds__(256) void edge_exp_kernel(
    const float* __restrict__ logits, const unsigned* __restrict__ m_ord,
    const int64_t* __restrict__ ei, float* __restrict__ aexp,
    float* __restrict__ denom) {
  const int t = blockIdx.x * 256 + threadIdx.x;
  if (t >= E_ * 4) return;
  const int e = t >> 2, h = t & 3;
  const int dst = (int)ei[E_ + e];
  float m = ord2f(m_ord[(size_t)dst * 4 + h]);
  if (!isfinite(m)) m = 0.0f;
  const float a = __expf(logits[t] - m);
  aexp[t] = a;
  atomicAdd(&denom[(size_t)dst * 4 + h], a);
}

// ---------- edge pass 3: weighted message scatter ----------
__global__ __launch_bounds__(256) void edge_scatter_kernel(
    const float* __restrict__ v, const float* __restrict__ aexp,
    const float* __restrict__ denom, const int64_t* __restrict__ ei,
    const float* __restrict__ ea, const float* __restrict__ Wel,
    float* __restrict__ out) {
  const int gw = (blockIdx.x * 256 + threadIdx.x) >> 5;
  const int lane = threadIdx.x & 31;
  const int e = gw >> 2, h = gw & 3;
  if (e >= E_) return;
  const int src = (int)ei[e];
  const int dst = (int)ei[E_ + e];
  const float alpha = aexp[(size_t)e * 4 + h] / (denom[(size_t)dst * 4 + h] + 1e-16f);
  const int co = h * 128 + lane * 4;
  const float4 vv = *(const float4*)(v + (size_t)src * HC + co);
  const float4 wv = *(const float4*)(Wel + co);
  const float ew = ea[e];
  float* op = out + (size_t)dst * HC + co;
  atomicAdd(op + 0, (vv.x + ew * wv.x) * alpha);
  atomicAdd(op + 1, (vv.y + ew * wv.y) * alpha);
  atomicAdd(op + 2, (vv.z + ew * wv.z) * alpha);
  atomicAdd(op + 3, (vv.w + ew * wv.w) * alpha);
}

// ---------- GraphNorm: per-channel sum / sumsq (16 rows per block) ----------
__global__ __launch_bounds__(256) void gn_partial_kernel(
    const float* __restrict__ out, float* __restrict__ gsum, float* __restrict__ gsq) {
  const int r0 = blockIdx.x * 16;
  const int c0 = threadIdx.x; // channels c0 and c0+256
  float s0 = 0.f, s1 = 0.f, q0 = 0.f, q1 = 0.f;
#pragma unroll 4
  for (int r = 0; r < 16; ++r) {
    const float* row = out + (size_t)(r0 + r) * HC;
    const float a = row[c0], b = row[c0 + 256];
    s0 += a; q0 += a * a;
    s1 += b; q1 += b * b;
  }
  atomicAdd(&gsum[c0], s0);        atomicAdd(&gsq[c0], q0);
  atomicAdd(&gsum[c0 + 256], s1);  atomicAdd(&gsq[c0 + 256], q1);
}

__global__ void gn_finalize_kernel(const float* __restrict__ gsum, const float* __restrict__ gsq,
                                   const float* __restrict__ gms,
                                   float* __restrict__ gmean, float* __restrict__ grstd) {
  const int c = threadIdx.x;
  const float mean = gsum[c] * (1.0f / N_);
  const float msq  = gsq[c]  * (1.0f / N_);
  const float ms = gms[c];
  // var of (out - ms*mean): E[x^2] - (2ms - ms^2) * mean^2
  const float var = msq - (2.0f * ms - ms * ms) * mean * mean;
  gmean[c] = mean;
  grstd[c] = rsqrtf(var + 1e-5f);
}

__global__ __launch_bounds__(256) void gn_normalize_kernel(
    const float* __restrict__ out, const float* __restrict__ gmean,
    const float* __restrict__ grstd, const float* __restrict__ gw,
    const float* __restrict__ gb, const float* __restrict__ gms,
    unsigned short* __restrict__ hb) {
  const int i = blockIdx.x * 256 + threadIdx.x; // over Kp*HC
  if (i >= Kp * HC) return;
  const int n = i >> 9, c = i & 511;
  unsigned short val = 0;
  if (n < N_) {
    const float sub = out[i] - gms[c] * gmean[c];
    float y = gw[c] * sub * grstd[c] + gb[c];
    y = fmaxf(y, 0.0f); // relu
    val = f2bf(y);
  }
  hb[i] = val;
}

// ---------- global min/max of G ----------
__global__ void minmax_init_kernel(unsigned* mm) { mm[0] = 0xFFFFFFFFu; mm[1] = 0u; }

__global__ __launch_bounds__(256) void minmax_kernel(const float* __restrict__ G,
                                                     unsigned* __restrict__ mm) {
  const int i = blockIdx.x * 256 + threadIdx.x; // 512*512 elements exactly
  const unsigned o = f2ord(G[i]);
  unsigned omin = o, omax = o;
#pragma unroll
  for (int off = 16; off > 0; off >>= 1) {
    const unsigned tmin = (unsigned)__shfl_xor((int)omin, off, 32);
    const unsigned tmax = (unsigned)__shfl_xor((int)omax, off, 32);
    omin = (tmin < omin) ? tmin : omin;
    omax = (tmax > omax) ? tmax : omax;
  }
  if ((threadIdx.x & 31) == 0) {
    atomicMin(&mm[0], omin);
    atomicMax(&mm[1], omax);
  }
}

// ---------- normalized upper-triangle extraction ----------
__global__ __launch_bounds__(256) void triu_kernel(const float* __restrict__ G,
                                                   const unsigned* __restrict__ mm,
                                                   float* __restrict__ outv) {
  const int i = blockIdx.x * 256 + threadIdx.x; // 512*512
  const int r = i >> 9, c = i & 511;
  if (c <= r) return;
  const float mn = ord2f(mm[0]);
  const float mx = ord2f(mm[1]);
  const float inv = 1.0f / (mx - mn + 1e-8f);
  const int idx = r * 511 - (r * (r - 1)) / 2 + (c - r - 1);
  outv[idx] = (G[i] - mn) * inv;
}

// ---------- host launcher ----------
extern "C" void kernel_launch(void* const* d_in, const int* in_sizes, int n_in,
                              void* d_out, int out_size, void* d_ws, size_t ws_size,
                              hipStream_t stream) {
  const float*   x   = (const float*)d_in[0];
  const int64_t* ei  = (const int64_t*)d_in[1];
  const float*   ea  = (const float*)d_in[2];
  const float*   Wq  = (const float*)d_in[3];
  const float*   bq  = (const float*)d_in[4];
  const float*   Wk  = (const float*)d_in[5];
  const float*   bk  = (const float*)d_in[6];
  const float*   Wv  = (const float*)d_in[7];
  const float*   bv  = (const float*)d_in[8];
  const float*   We  = (const float*)d_in[9];
  const float*   Ws  = (const float*)d_in[10];
  const float*   bs  = (const float*)d_in[11];
  const float*   gw  = (const float*)d_in[12];
  const float*   gb  = (const float*)d_in[13];
  const float*   gms = (const float*)d_in[14];

  char* ws = (char*)d_ws;
  size_t off = 0;
  auto alloc = [&](size_t bytes) -> char* {
    char* p = ws + off;
    off = (off + bytes + 255) & ~(size_t)255;
    return p;
  };
  unsigned short* hb   = (unsigned short*)alloc((size_t)Kp * HC * 2);
  unsigned short* wqb  = (unsigned short*)alloc((size_t)2 * 512 * 512 * 2);
  unsigned short* wkb  = (unsigned short*)alloc((size_t)2 * 512 * 512 * 2);
  unsigned short* wvb  = (unsigned short*)alloc((size_t)2 * 512 * 512 * 2);
  unsigned short* wsb  = (unsigned short*)alloc((size_t)2 * 512 * 512 * 2);
  float* q      = (float*)alloc((size_t)N_ * HC * 4);
  float* k      = (float*)alloc((size_t)N_ * HC * 4);
  float* v      = (float*)alloc((size_t)N_ * HC * 4);
  float* sout   = (float*)alloc((size_t)N_ * HC * 4); // skip + message accumulation
  float* logits = (float*)alloc((size_t)E_ * 4 * 4);
  float* aexp   = (float*)alloc((size_t)E_ * 4 * 4);
  unsigned* m_ord = (unsigned*)alloc((size_t)N_ * 4 * 4);
  float* denom    = (float*)alloc((size_t)N_ * 4 * 4);
  float* gsum   = (float*)alloc(512 * 4);
  float* gsq    = (float*)alloc(512 * 4);
  float* gmean  = (float*)alloc(512 * 4);
  float* grstd  = (float*)alloc(512 * 4);
  float* G      = (float*)alloc((size_t)512 * 512 * 4);
  unsigned* mm  = (unsigned*)alloc(2 * 4);

  // convert inputs to bf16 (h padded to Kp rows with zeros)
  f32_to_bf16_kernel<<<(Kp * HC) / 256, 256, 0, stream>>>(x, hb, Kp * HC, N_ * HC);
  const int wcnt = 2 * 512 * 512;
  f32_to_bf16_kernel<<<(wcnt + 255) / 256, 256, 0, stream>>>(Wq, wqb, wcnt, wcnt);
  f32_to_bf16_kernel<<<(wcnt + 255) / 256, 256, 0, stream>>>(Wk, wkb, wcnt, wcnt);
  f32_to_bf16_kernel<<<(wcnt + 255) / 256, 256, 0, stream>>>(Wv, wvb, wcnt, wcnt);
  f32_to_bf16_kernel<<<(wcnt + 255) / 256, 256, 0, stream>>>(Ws, wsb, wcnt, wcnt);

  for (int l = 0; l < 2; ++l) {
    hipMemsetAsync(m_ord, 0, (size_t)N_ * 4 * 4, stream); // ord(-NaN): identity for max, decodes non-finite
    hipMemsetAsync(denom, 0, (size_t)N_ * 4 * 4, stream);
    hipMemsetAsync(gsum, 0, 512 * 4, stream);
    hipMemsetAsync(gsq, 0, 512 * 4, stream);

    Gemm4 p;
    p.B[0] = wqb + (size_t)l * 512 * 512; p.bias[0] = bq + l * 512; p.C[0] = q;
    p.B[1] = wkb + (size_t)l * 512 * 512; p.bias[1] = bk + l * 512; p.C[1] = k;
    p.B[2] = wvb + (size_t)l * 512 * 512; p.bias[2] = bv + l * 512; p.C[2] = v;
    p.B[3] = wsb + (size_t)l * 512 * 512; p.bias[3] = bs + l * 512; p.C[3] = sout;
    gemm4_bf16_kernel<<<dim3(N_ / 16, 4), 256, 0, stream>>>(hb, p, HC);

    const float* Wel = We + (size_t)l * 512;
    edge_logits_kernel<<<(E_ * 4) / 8, 256, 0, stream>>>(q, k, ei, ea, Wel, logits, m_ord);
    edge_exp_kernel<<<(E_ * 4 + 255) / 256, 256, 0, stream>>>(logits, m_ord, ei, aexp, denom);
    edge_scatter_kernel<<<(E_ * 4) / 8, 256, 0, stream>>>(v, aexp, denom, ei, ea, Wel, sout);

    gn_partial_kernel<<<N_ / 16, 256, 0, stream>>>(sout, gsum, gsq);
    gn_finalize_kernel<<<1, 512, 0, stream>>>(gsum, gsq, gms + l * 512, gmean, grstd);
    gn_normalize_kernel<<<(Kp * HC) / 256, 256, 0, stream>>>(
        sout, gmean, grstd, gw + l * 512, gb + l * 512, gms + l * 512, hb);
  }

  gram_bf16_kernel<<<128, 256, 0, stream>>>(hb, G, Kp);
  minmax_init_kernel<<<1, 1, 0, stream>>>(mm);
  minmax_kernel<<<(512 * 512) / 256, 256, 0, stream>>>(G, mm);
  triu_kernel<<<(512 * 512) / 256, 256, 0, stream>>>(G, mm, (float*)d_out);
}
